// DenseVanillaAttention_16286515986790
// MI455X (gfx1250) — matrix operand order repaired and checked by reading, not compiled, not run on hardware
//
#include <hip/hip_runtime.h>
#include <hip/hip_bf16.h>

// CDNA5 / gfx1250 — wave32, WMMA 16x16x32 bf16 (f32 accum), async global->LDS.

typedef __attribute__((ext_vector_type(16))) __bf16 v16bf;
typedef __attribute__((ext_vector_type(8)))  float  v8f;
typedef int v4i_gcc __attribute__((vector_size(16)));   // matches builtin param type

#define D_DIM 512
#define S_DIM 1024
#define B_DIM 32
#define H_DIM 64

// LDS tile pad stride in dwords: 80 bytes/row -> 16B-aligned rows for B128
// async stores, and 20*row mod 64 is distinct over any 16 rows (no bank
// conflicts in the fragment gathers).
#define LDST 20

union BF16Frag { v16bf v; unsigned int u[8]; };

__device__ __forceinline__ unsigned int pack_f32(float lo, float hi) {
    unsigned short a = __builtin_bit_cast(unsigned short, (__bf16)lo);
    unsigned short b = __builtin_bit_cast(unsigned short, (__bf16)hi);
    return ((unsigned int)b << 16) | (unsigned int)a;
}
__device__ __forceinline__ unsigned int pack_bf(__bf16 lo, __bf16 hi) {
    unsigned short a = __builtin_bit_cast(unsigned short, lo);
    unsigned short b = __builtin_bit_cast(unsigned short, hi);
    return ((unsigned int)b << 16) | (unsigned int)a;
}

// ---------------------------------------------------------------------------
// Async global -> LDS (ASYNCcnt path), 128-bit per lane.
// Builtin signature (from hipcc diagnostic): param0 is v4i addrspace(1)*.
// ---------------------------------------------------------------------------
#if defined(__has_builtin)
#if __has_builtin(__builtin_amdgcn_global_load_async_to_lds_b128)
#define HAVE_ASYNC_B128 1
#endif
#if __has_builtin(__builtin_amdgcn_s_wait_asynccnt)
#define HAVE_WAIT_ASYNC 1
#endif
#endif

__device__ __forceinline__ void async_copy_b128(const void* g, const void* l) {
    // Flat LDS addresses carry the LDS byte offset in bits [31:0] (ISA 10.2).
    unsigned long long ga = (unsigned long long)g;
    unsigned int       la = (unsigned int)(unsigned long long)l;
#ifdef HAVE_ASYNC_B128
    __builtin_amdgcn_global_load_async_to_lds_b128(
        (__attribute__((address_space(1))) v4i_gcc*)ga,
        (__attribute__((address_space(3))) v4i_gcc*)la, 0, 0);
#else
    asm volatile("global_load_async_to_lds_b128 %0, %1, off"
                 :: "v"(la), "v"(ga) : "memory");
#endif
}
__device__ __forceinline__ void wait_async_0() {
#ifdef HAVE_WAIT_ASYNC
    __builtin_amdgcn_s_wait_asynccnt(0);
#else
    asm volatile("s_wait_asynccnt 0x0" ::: "memory");
#endif
}
__device__ __forceinline__ void wait_async_3() {
#ifdef HAVE_WAIT_ASYNC
    __builtin_amdgcn_s_wait_asynccnt(3);
#else
    asm volatile("s_wait_asynccnt 0x3" ::: "memory");
#endif
}

// ---------------------------------------------------------------------------
// One K=32 WMMA step, wave owns two 16x16 C tiles.
// LDS layouts (dword units, stride LDST):
//   As[row][kp] : 64 rows x 16 kpairs (M x K, row-major pairs)
//   Bs[n][kp]   : cols x 16 kpairs (B stored transposed: N x K)
// Fragment gathers (contiguous k order, same for A and B; not the ISA table order):
//   A lane L: M=L%16; VGPR v: K = 2v + (L>=16?16:0)             -> kpair v+hi*8
//   B lane L: N=L%16; VGPR v: K = 2v + (L>=16?16:0)             -> kpair v+hi*8
// ---------------------------------------------------------------------------
__device__ __forceinline__ void wmma_pair(const unsigned int* As, const unsigned int* Bs,
                                          int wm, int wn, int lr, int hi,
                                          v8f& acc0, v8f& acc1)
{
    BF16Frag a, b0, b1;
#pragma unroll
    for (int v = 0; v < 8; ++v) {
        a.u[v]  = As[(wm * 16 + lr) * LDST + (v + hi * 8)];
        b0.u[v] = Bs[(wn * 32 + lr) * LDST + (v + hi * 8)];
        b1.u[v] = Bs[(wn * 32 + 16 + lr) * LDST + (v + hi * 8)];
    }
    acc0 = __builtin_amdgcn_wmma_f32_16x16x32_bf16(false, a.v, false, b0.v,
                                                   (short)0, acc0, false, false);
    acc1 = __builtin_amdgcn_wmma_f32_16x16x32_bf16(false, a.v, false, b1.v,
                                                   (short)0, acc1, false, false);
}

// ---------------------------------------------------------------------------
// K0: fp32 -> bf16 convert (w2 weights)
// ---------------------------------------------------------------------------
__global__ void cvt_bf16_kernel(const float* __restrict__ in, __bf16* __restrict__ out, int n) {
    int i = blockIdx.x * blockDim.x + threadIdx.x;
    if (i < n) out[i] = (__bf16)in[i];
}

// ---------------------------------------------------------------------------
// K1: projection GEMM. out_bf16[M,N] = act(A_f32[M,K] @ B_f32[K,N] + bias)
// Grid: (N/64, M/64). 256 threads = 8 waves, 4x2 wave tiling, 64x64 block tile.
// ---------------------------------------------------------------------------
__global__ __launch_bounds__(256) void gemm_proj_kernel(
    const float* __restrict__ A, const float* __restrict__ B,
    const float* __restrict__ bias, __bf16* __restrict__ out,
    int K, int lda, int ldb, int ldo, int relu)
{
    __shared__ __align__(16) unsigned int As[64 * LDST];
    __shared__ __align__(16) unsigned int Bs[64 * LDST];

    const int m0 = blockIdx.y * 64;
    const int n0 = blockIdx.x * 64;
    const int t = threadIdx.x;
    const int lane = t & 31, wid = t >> 5;
    const int wm = wid & 3, wn = wid >> 2;
    const int lr = lane & 15, hi = lane >> 4;

    v8f acc0 = {}, acc1 = {};

    for (int kk0 = 0; kk0 < K; kk0 += 32) {
#pragma unroll
        for (int i = 0; i < 4; ++i) {
            int idx = t + i * 256;
            int row = idx >> 4, kp = idx & 15;
            const float* ap = A + (size_t)(m0 + row) * lda + kk0 + kp * 2;
            As[row * LDST + kp] = pack_f32(ap[0], ap[1]);
            int n = idx & 63, kq = idx >> 6;
            const float* bp = B + (size_t)(kk0 + kq * 2) * ldb + n0 + n;
            Bs[n * LDST + kq] = pack_f32(bp[0], bp[ldb]);
        }
        if (kk0 + 32 < K) {
            __builtin_prefetch(A + (size_t)(m0 + (t >> 4)) * lda + kk0 + 32 + ((t & 15) * 2), 0, 1);
        }
        __syncthreads();
        wmma_pair(As, Bs, wm, wn, lr, hi, acc0, acc1);
        __syncthreads();
    }

#pragma unroll
    for (int r = 0; r < 8; ++r) {
        int row = m0 + wm * 16 + hi * 8 + r;
        int c0 = n0 + wn * 32 + lr;
        float x0 = acc0[r] + bias[c0];
        float x1 = acc1[r] + bias[c0 + 16];
        if (relu) { x0 = fmaxf(x0, 0.0f); x1 = fmaxf(x1, 0.0f); }
        out[(size_t)row * ldo + c0] = (__bf16)x0;
        out[(size_t)row * ldo + c0 + 16] = (__bf16)x1;
    }
}

// ---------------------------------------------------------------------------
// K2: scores[b,s,t] = h[b,s,:] @ w2[:,t] + w2_b[t] + q[b,s,:] . k[b,t,:]
// Block tile 64 (s) x 128 (t); 8 waves as 4 (M) x 2 (N), each wave 16x64 (4 WMMA
// per A fragment). Phase 2 (K=512, bf16 operands) uses double-buffered async
// global->LDS B128 copies. Grid: (S/128, S/64, B). Output fp32 (pre-softmax).
// ---------------------------------------------------------------------------
__global__ __launch_bounds__(256) void scores_kernel(
    const __bf16* __restrict__ qb, const __bf16* __restrict__ kb,
    const __bf16* __restrict__ hb, const __bf16* __restrict__ w2b,
    const float* __restrict__ w2_bias, float* __restrict__ scores)
{
    __shared__ __align__(16) unsigned int As[2][64 * LDST];
    __shared__ __align__(16) unsigned int Bs[2][128 * LDST];

    const int b = blockIdx.z;
    const int s0 = blockIdx.y * 64;    // rows (s)
    const int t0 = blockIdx.x * 128;   // cols (t)
    const int t = threadIdx.x;
    const int lane = t & 31, wid = t >> 5;
    const int wm = wid & 3, wn = wid >> 2;
    const int lr = lane & 15, hi = lane >> 4;

    v8f acc[4] = {{}, {}, {}, {}};

    // ---- Phase 1: MLP term, K = 64. A = h (bf16), B = w2 (bf16) transposed.
    const __bf16* hrow = hb + (size_t)b * S_DIM * H_DIM;
    for (int kk0 = 0; kk0 < H_DIM; kk0 += 32) {
#pragma unroll
        for (int i = 0; i < 4; ++i) {       // A tile: 1024 dwords
            int idx = t + i * 256;
            int row = idx >> 4, kp = idx & 15;
            As[0][row * LDST + kp] =
                ((const unsigned int*)(hrow + (size_t)(s0 + row) * H_DIM + kk0))[kp];
        }
#pragma unroll
        for (int i = 0; i < 8; ++i) {       // B tile: 2048 dwords (128 cols)
            int idx = t + i * 256;
            int n = idx & 127, kq = idx >> 7;
            const __bf16* wp = w2b + (size_t)(kk0 + kq * 2) * S_DIM + t0 + n;
            Bs[0][n * LDST + kq] = pack_bf(wp[0], wp[S_DIM]);
        }
        __syncthreads();
        {
            BF16Frag a;
#pragma unroll
            for (int v = 0; v < 8; ++v)
                a.u[v] = As[0][(wm * 16 + lr) * LDST + (v + hi * 8)];
#pragma unroll
            for (int j = 0; j < 4; ++j) {
                BF16Frag bf;
#pragma unroll
                for (int v = 0; v < 8; ++v)
                    bf.u[v] = Bs[0][(wn * 64 + j * 16 + lr) * LDST + (v + hi * 8)];
                acc[j] = __builtin_amdgcn_wmma_f32_16x16x32_bf16(
                    false, a.v, false, bf.v, (short)0, acc[j], false, false);
            }
        }
        __syncthreads();
    }

    // ---- Phase 2: q @ k^T, K = 512. Pure bf16 byte copies -> async to LDS,
    // double buffered. B = k^T, so the transposed-LDS tile is contiguous rows
    // of k. 3 async B128 ops per thread per tile (A:1, B:2).
    const __bf16* qrow = qb + (size_t)b * S_DIM * D_DIM;
    const __bf16* krow = kb + (size_t)b * S_DIM * D_DIM;

    auto issue_tile = [&](int buf, int kk0) {
        {   // A tile: 64 rows x 16 dwords; lane -> row t>>2, dword group (t&3)*4
            int row = t >> 2, kg = (t & 3) * 4;
            async_copy_b128(qrow + (size_t)(s0 + row) * D_DIM + kk0 + kg * 2,
                            &As[buf][row * LDST + kg]);
        }
#pragma unroll
        for (int j = 0; j < 2; ++j) {   // B tile: 128 rows x 16 dwords
            int idx = t + j * 256;
            int row = idx >> 2, kg = (idx & 3) * 4;
            async_copy_b128(krow + (size_t)(t0 + row) * D_DIM + kk0 + kg * 2,
                            &Bs[buf][row * LDST + kg]);
        }
    };

    issue_tile(0, 0);
    for (int kk0 = 0; kk0 < D_DIM; kk0 += 32) {
        int cur = (kk0 >> 5) & 1;
        if (kk0 + 32 < D_DIM) {
            issue_tile(cur ^ 1, kk0 + 32);
            wait_async_3();             // oldest tile (cur) complete
        } else {
            wait_async_0();
        }
        __syncthreads();
        {
            BF16Frag a;
#pragma unroll
            for (int v = 0; v < 8; ++v)
                a.u[v] = As[cur][(wm * 16 + lr) * LDST + (v + hi * 8)];
#pragma unroll
            for (int j = 0; j < 4; ++j) {
                BF16Frag bf;
#pragma unroll
                for (int v = 0; v < 8; ++v)
                    bf.u[v] = Bs[cur][(wn * 64 + j * 16 + lr) * LDST + (v + hi * 8)];
                acc[j] = __builtin_amdgcn_wmma_f32_16x16x32_bf16(
                    false, a.v, false, bf.v, (short)0, acc[j], false, false);
            }
        }
        __syncthreads();
    }

    float* srow = scores + (size_t)b * S_DIM * S_DIM;
#pragma unroll
    for (int r = 0; r < 8; ++r) {
        int row = s0 + wm * 16 + hi * 8 + r;
        int cb = t0 + wn * 64 + lr;
#pragma unroll
        for (int j = 0; j < 4; ++j)
            srow[(size_t)row * S_DIM + cb + j * 16] = acc[j][r] + w2_bias[cb + j * 16];
    }
}

// ---------------------------------------------------------------------------
// K3: in-place row softmax over S=1024. One 256-thread block per row.
// ---------------------------------------------------------------------------
__global__ __launch_bounds__(256) void softmax_kernel(float* __restrict__ p)
{
    __shared__ float redm[8];
    __shared__ float reds[8];
    float* x = p + (size_t)blockIdx.x * S_DIM;
    const int t = threadIdx.x;
    const int lane = t & 31, w = t >> 5;

    float v[4];
    float m = -3.0e38f;
#pragma unroll
    for (int i = 0; i < 4; ++i) { v[i] = x[t + i * 256]; m = fmaxf(m, v[i]); }
#pragma unroll
    for (int off = 16; off > 0; off >>= 1) m = fmaxf(m, __shfl_xor(m, off, 32));
    if (lane == 0) redm[w] = m;
    __syncthreads();
    m = redm[0];
#pragma unroll
    for (int i = 1; i < 8; ++i) m = fmaxf(m, redm[i]);

    float s = 0.0f;
#pragma unroll
    for (int i = 0; i < 4; ++i) { v[i] = __expf(v[i] - m); s += v[i]; }
#pragma unroll
    for (int off = 16; off > 0; off >>= 1) s += __shfl_xor(s, off, 32);
    if (lane == 0) reds[w] = s;
    __syncthreads();
    s = reds[0];
#pragma unroll
    for (int i = 1; i < 8; ++i) s += reds[i];

    float inv = __frcp_rn(s);
#pragma unroll
    for (int i = 0; i < 4; ++i) x[t + i * 256] = v[i] * inv;
}

// ---------------------------------------------------------------------------
// K4: out[b] = attention[b] @ value[b]. A = attn fp32 (->bf16 on load),
// B = value fp32 (->bf16, transposed into LDS). Grid (D/64, S/64, B).
// ---------------------------------------------------------------------------
__global__ __launch_bounds__(256) void attnv_kernel(
    const float* __restrict__ attn, const float* __restrict__ value,
    float* __restrict__ out)
{
    __shared__ __align__(16) unsigned int As[64 * LDST];
    __shared__ __align__(16) unsigned int Bs[64 * LDST];

    const int b = blockIdx.z;
    const int m0 = blockIdx.y * 64;   // s rows
    const int n0 = blockIdx.x * 64;   // d cols
    const int t = threadIdx.x;
    const int lane = t & 31, wid = t >> 5;
    const int wm = wid & 3, wn = wid >> 2;
    const int lr = lane & 15, hi = lane >> 4;

    const float* A = attn + (size_t)b * S_DIM * S_DIM;
    const float* B = value + (size_t)b * S_DIM * D_DIM;

    v8f acc0 = {}, acc1 = {};
    for (int kk0 = 0; kk0 < S_DIM; kk0 += 32) {
#pragma unroll
        for (int i = 0; i < 4; ++i) {
            int idx = t + i * 256;
            int row = idx >> 4, kp = idx & 15;
            const float* ap = A + (size_t)(m0 + row) * S_DIM + kk0 + kp * 2;
            As[row * LDST + kp] = pack_f32(ap[0], ap[1]);
            int n = idx & 63, kq = idx >> 6;
            const float* bp = B + (size_t)(kk0 + kq * 2) * D_DIM + n0 + n;
            Bs[n * LDST + kq] = pack_f32(bp[0], bp[D_DIM]);
        }
        if (kk0 + 32 < S_DIM) {
            __builtin_prefetch(A + (size_t)(m0 + (t >> 4)) * S_DIM + kk0 + 32 + ((t & 15) * 2), 0, 1);
            __builtin_prefetch(B + (size_t)(kk0 + 32 + (t >> 6)) * D_DIM + n0 + (t & 63), 0, 1);
        }
        __syncthreads();
        wmma_pair(As, Bs, wm, wn, lr, hi, acc0, acc1);
        __syncthreads();
    }

    float* orow = out + (size_t)b * S_DIM * D_DIM;
#pragma unroll
    for (int r = 0; r < 8; ++r) {
        int row = m0 + wm * 16 + hi * 8 + r;
        int c0 = n0 + wn * 32 + lr;
        orow[(size_t)row * D_DIM + c0]      = acc0[r];
        orow[(size_t)row * D_DIM + c0 + 16] = acc1[r];
    }
}

// ---------------------------------------------------------------------------
extern "C" void kernel_launch(void* const* d_in, const int* in_sizes, int n_in,
                              void* d_out, int out_size, void* d_ws, size_t ws_size,
                              hipStream_t stream)
{
    (void)in_sizes; (void)n_in; (void)out_size; (void)ws_size;

    const float* query = (const float*)d_in[0];
    const float* key   = (const float*)d_in[1];
    const float* value = (const float*)d_in[2];
    const float* w1_w  = (const float*)d_in[3];
    const float* w1_b  = (const float*)d_in[4];
    const float* w2_w  = (const float*)d_in[5];
    const float* w2_b  = (const float*)d_in[6];
    const float* q_w   = (const float*)d_in[7];
    const float* q_b   = (const float*)d_in[8];
    const float* k_w   = (const float*)d_in[9];
    const float* k_b   = (const float*)d_in[10];

    float* outp  = (float*)d_out;                                  // [B,S,D]
    float* attnp = outp + (size_t)B_DIM * S_DIM * D_DIM;           // [B,S,S]

    char* ws = (char*)d_ws;
    __bf16* qbf  = (__bf16*)ws; ws += (size_t)B_DIM * S_DIM * D_DIM * 2;
    __bf16* kbf  = (__bf16*)ws; ws += (size_t)B_DIM * S_DIM * D_DIM * 2;
    __bf16* hbf  = (__bf16*)ws; ws += (size_t)B_DIM * S_DIM * H_DIM * 2;
    __bf16* w2bf = (__bf16*)ws; ws += (size_t)H_DIM * S_DIM * 2;

    // K0: w2 weights to bf16
    cvt_bf16_kernel<<<(H_DIM * S_DIM + 255) / 256, 256, 0, stream>>>(w2_w, w2bf, H_DIM * S_DIM);

    // K1: projections (A treated as [B*S, D])
    dim3 gq(D_DIM / 64, (B_DIM * S_DIM) / 64);
    gemm_proj_kernel<<<gq, 256, 0, stream>>>(query, q_w, q_b, qbf, D_DIM, D_DIM, D_DIM, D_DIM, 0);
    gemm_proj_kernel<<<gq, 256, 0, stream>>>(key,   k_w, k_b, kbf, D_DIM, D_DIM, D_DIM, D_DIM, 0);
    dim3 gh(H_DIM / 64, (B_DIM * S_DIM) / 64);
    gemm_proj_kernel<<<gh, 256, 0, stream>>>(query, w1_w, w1_b, hbf, D_DIM, D_DIM, H_DIM, H_DIM, 1);

    // K2: pre-softmax scores -> attention region of d_out (async double-buffered)
    dim3 gs(S_DIM / 128, S_DIM / 64, B_DIM);
    scores_kernel<<<gs, 256, 0, stream>>>(qbf, kbf, hbf, w2bf, w2_b, attnp);

    // K3: in-place softmax
    softmax_kernel<<<B_DIM * S_DIM, 256, 0, stream>>>(attnp);

    // K4: out = attention @ value
    dim3 go(D_DIM / 64, S_DIM / 64, B_DIM);
    attnv_kernel<<<go, 256, 0, stream>>>(attnp, value, outp);
}
